// MultiheadAttention_8091718385882
// MI455X (gfx1250) — compile-verified, matrix operand order checked by reading, not run on hardware
//
#include <hip/hip_runtime.h>

typedef __attribute__((ext_vector_type(16))) __bf16 v16bf;
typedef __attribute__((ext_vector_type(2)))  __bf16 v2bf;
typedef __attribute__((ext_vector_type(8)))  float  v8f;

#define Bc  4
#define Sc  2048
#define Dc  1024
#define Hc  16
#define HDc 64

// f32 -> bf16, round-to-nearest-even, integer fallback
static __device__ inline __bf16 f2bf(float f) {
  union { float f; unsigned u; } a; a.f = f;
  unsigned r = a.u + 0x7FFFu + ((a.u >> 16) & 1u);
  union { unsigned short s; __bf16 b; } o;
  o.s = (unsigned short)(r >> 16);
  return o.b;
}

// packed f32x2 -> bf16x2 (native on CDNA5 if the builtin is declared)
static __device__ inline v2bf pk2(float a, float b) {
#if __has_builtin(__builtin_amdgcn_cvt_pk_bf16_f32)
  return __builtin_amdgcn_cvt_pk_bf16_f32(a, b);
#else
  v2bf r; r[0] = f2bf(a); r[1] = f2bf(b); return r;
#endif
}

// 16-lane XOR-pattern max reduction (rows live in 16-lane half-waves).
static __device__ inline float rowmax16(float t) {
#if __has_builtin(__builtin_amdgcn_ds_swizzle)
  t = fmaxf(t, __builtin_bit_cast(float, __builtin_amdgcn_ds_swizzle(__builtin_bit_cast(int, t), 0x041F)));
  t = fmaxf(t, __builtin_bit_cast(float, __builtin_amdgcn_ds_swizzle(__builtin_bit_cast(int, t), 0x081F)));
  t = fmaxf(t, __builtin_bit_cast(float, __builtin_amdgcn_ds_swizzle(__builtin_bit_cast(int, t), 0x101F)));
  t = fmaxf(t, __builtin_bit_cast(float, __builtin_amdgcn_ds_swizzle(__builtin_bit_cast(int, t), 0x201F)));
#else
  #pragma unroll
  for (int m = 1; m < 16; m <<= 1) t = fmaxf(t, __shfl_xor(t, m, 32));
#endif
  return t;
}

// Async memory -> LDS copy of 16 bytes per lane (tracked on ASYNCcnt).
static __device__ inline void async_ld16(unsigned lds_off, const void* gptr) {
  asm volatile("global_load_async_to_lds_b128 %0, %1, off"
               :: "v"(lds_off), "v"(gptr) : "memory");
}
static __device__ inline void wait_async0() {
  asm volatile("s_wait_asynccnt 0x0" ::: "memory");
}
static __device__ inline void wait_async2() {   // allow the 2 newest (next tile) in flight
  asm volatile("s_wait_asynccnt 0x2" ::: "memory");
}

// Load a 16x32 bf16 WMMA fragment (A layout; B loaded as A of the transpose).
// lane<16:  row = row0+lane,    k = {0..7, 16..23}
// lane>=16: row = row0+lane-16, k = {8..15, 24..31}
static __device__ inline v16bf ld_frag(const __bf16* base, int stride, int row0) {
  const int lane = threadIdx.x & 31;
  const __bf16* p = base + (size_t)(row0 + (lane & 15)) * stride + ((lane >> 4) << 3);
  union { v16bf v; uint4 u[2]; } f;
  f.u[0] = *(const uint4*)(p);
  f.u[1] = *(const uint4*)(p + 16);
  return f.v;
}

static __device__ inline v16bf ones_frag() {
  union { v16bf v; unsigned short s[16]; } t;
  #pragma unroll
  for (int i = 0; i < 16; ++i) t.s[i] = 0x3F80;   // bf16 1.0
  return t.v;
}

// ---------------------------------------------------------------------------
// GEMM: C[M,N] = A[M,K] * W[N,K]^T  (M = 8192, N = K = 1024)
// Block tile 128x128, BK=32, register-prefetch pipeline: global loads for
// tile k+1 issue before the 8 WMMAs of tile k. 8 waves (4x2), 32x64 each.
// MODE 0: f32 [M,N] -> d_out.  MODE 1: bf16 [B,H,S,HD].  MODE 2: bf16 [B,H,HD,S].
// ---------------------------------------------------------------------------
template<int MODE, bool ABF16>
__global__ __launch_bounds__(256) void gemm_wmma(const void* __restrict__ Ain,
                                                 const float* __restrict__ W,
                                                 void* __restrict__ Out,
                                                 float scale) {
  __shared__ __align__(16) __bf16 As[128 * 32];
  __shared__ __align__(16) __bf16 Bs[128 * 32];

  const int tid  = threadIdx.x;
  const int lane = tid & 31;
  const int w    = tid >> 5;
  const int wm   = (w >> 1) * 32;
  const int wn   = (w & 1) * 64;
  const int m0   = blockIdx.y * 128;
  const int n0   = blockIdx.x * 128;

  const int sr = tid >> 1, sc = (tid & 1) * 16;   // staging: 16 elems/thread
  const float*  aF = (const float*) Ain + (size_t)(m0 + sr) * Dc + sc;
  const __bf16* aB = (const __bf16*)Ain + (size_t)(m0 + sr) * Dc + sc;
  const float*  wF = W + (size_t)(n0 + sr) * Dc + sc;
  __bf16* dstA = As + sr * 32 + sc;
  __bf16* dstB = Bs + sr * 32 + sc;

  v8f acc[2][4] = {};
  float4 aR[4], wR[4];
  uint4  aH[2];

  // prologue: prefetch k0 = 0
  if (ABF16) {
    aH[0] = ((const uint4*)aB)[0];
    aH[1] = ((const uint4*)aB)[1];
  } else {
    #pragma unroll
    for (int i = 0; i < 4; ++i) aR[i] = ((const float4*)aF)[i];
  }
  #pragma unroll
  for (int i = 0; i < 4; ++i) wR[i] = ((const float4*)wF)[i];

  for (int k0 = 0; k0 < Dc; k0 += 32) {
    // store phase: convert prefetched regs -> LDS
    if (ABF16) {
      ((uint4*)dstA)[0] = aH[0]; ((uint4*)dstA)[1] = aH[1];
    } else {
      union { v2bf p[8]; uint4 u[2]; } t;
      const float* f = (const float*)aR;
      #pragma unroll
      for (int i = 0; i < 8; ++i) t.p[i] = pk2(f[2 * i], f[2 * i + 1]);
      ((uint4*)dstA)[0] = t.u[0]; ((uint4*)dstA)[1] = t.u[1];
    }
    {
      union { v2bf p[8]; uint4 u[2]; } t;
      const float* f = (const float*)wR;
      #pragma unroll
      for (int i = 0; i < 8; ++i) t.p[i] = pk2(f[2 * i], f[2 * i + 1]);
      ((uint4*)dstB)[0] = t.u[0]; ((uint4*)dstB)[1] = t.u[1];
    }
    __syncthreads();

    // prefetch next tile (overlaps with WMMA below)
    if (k0 + 32 < Dc) {
      if (ABF16) {
        aH[0] = ((const uint4*)(aB + k0 + 32))[0];
        aH[1] = ((const uint4*)(aB + k0 + 32))[1];
      } else {
        #pragma unroll
        for (int i = 0; i < 4; ++i) aR[i] = ((const float4*)(aF + k0 + 32))[i];
      }
      #pragma unroll
      for (int i = 0; i < 4; ++i) wR[i] = ((const float4*)(wF + k0 + 32))[i];
    }

    v16bf a[2], bfrag[4];
    #pragma unroll
    for (int mt = 0; mt < 2; ++mt) a[mt] = ld_frag(As, 32, wm + mt * 16);
    #pragma unroll
    for (int nt = 0; nt < 4; ++nt) bfrag[nt] = ld_frag(Bs, 32, wn + nt * 16);
    #pragma unroll
    for (int mt = 0; mt < 2; ++mt)
      #pragma unroll
      for (int nt = 0; nt < 4; ++nt)
        acc[mt][nt] = __builtin_amdgcn_wmma_f32_16x16x32_bf16(
            false, a[mt], false, bfrag[nt], (short)0, acc[mt][nt], false, false);
    __syncthreads();
  }

  const int half = lane >> 4;
  #pragma unroll
  for (int mt = 0; mt < 2; ++mt)
    #pragma unroll
    for (int nt = 0; nt < 4; ++nt)
      #pragma unroll
      for (int r = 0; r < 8; ++r) {
        const int gm = m0 + wm + mt * 16 + r + half * 8;   // C layout: M = vgpr + 8*half
        const int gn = n0 + wn + nt * 16 + (lane & 15);    //           N = lane % 16
        const float v = acc[mt][nt][r] * scale;
        if (MODE == 0) {
          ((float*)Out)[(size_t)gm * Dc + gn] = v;
        } else {
          const int b  = gm >> 11, s  = gm & (Sc - 1);
          const int h  = gn >> 6,  hd = gn & (HDc - 1);
          const size_t idx = (MODE == 1)
              ? ((size_t)((b * Hc + h) * Sc + s) * HDc + hd)
              : ((size_t)((b * Hc + h) * HDc + hd) * Sc + s);
          ((__bf16*)Out)[idx] = f2bf(v);
        }
      }
}

// ---------------------------------------------------------------------------
// Causal flash attention. Grid: (S/128, B*H). 256 threads = 8 waves.
// Wave w owns q rows [q0+16w, q0+16w+16). 32-key K/V tiles double-buffered in
// LDS via async loads: issue tile j+1, s_wait_asynccnt 2, compute tile j.
// Qp,Kp: [B,H,S,HD] bf16 (Q pre-scaled).  Vt: [B,H,HD,S] bf16.
// ---------------------------------------------------------------------------
__global__ __launch_bounds__(256) void flash_attn(const __bf16* __restrict__ Qp,
                                                  const __bf16* __restrict__ Kp,
                                                  const __bf16* __restrict__ Vt,
                                                  __bf16* __restrict__ Oout) {
  __shared__ __align__(16) __bf16 Ks[2][32 * 64];    // [buf][key][d]
  __shared__ __align__(16) __bf16 Vs[2][64 * 32];    // [buf][d][key]
  __shared__ __align__(16) __bf16 Ps[8 * 16 * 32];   // per-wave P bounce

  const int tid  = threadIdx.x;
  const int lane = tid & 31;
  const int w    = tid >> 5;
  const int half = lane >> 4;
  const int bh   = blockIdx.y;                 // b*H + h
  const int b    = bh >> 4, h = bh & 15;
  const int q0   = blockIdx.x * 128;
  const int qw   = q0 + w * 16;

  // Q: two 16x32 A-fragments (d = 0..31, 32..63).
  const __bf16* Qbase = Qp + ((size_t)bh * Sc + qw) * HDc;
  v16bf qf[2];
  #pragma unroll
  for (int f = 0; f < 2; ++f) {
    const __bf16* p = Qbase + (size_t)(lane & 15) * HDc + f * 32 + half * 8;
    union { v16bf v; uint4 u[2]; } t;
    t.u[0] = *(const uint4*)(p);
    t.u[1] = *(const uint4*)(p + 16);
    qf[f] = t.v;
  }
  const v16bf onesf = ones_frag();

  v8f o[4] = {};
  float mrow[8], lrow[8];
  #pragma unroll
  for (int r = 0; r < 8; ++r) { mrow[r] = -1e30f; lrow[r] = 0.0f; }

  const int ntiles = (q0 >> 5) + 4;            // causal upper bound for block
  const int kr = tid >> 3, kc = (tid & 7) * 8; // K staging: [32][64], 8 elems/thread
  const int vd = tid >> 2, vc = (tid & 3) * 8; // V staging: [64][32]
  const unsigned lK[2] = { (unsigned)(uintptr_t)(Ks[0] + kr * 64 + kc),
                           (unsigned)(uintptr_t)(Ks[1] + kr * 64 + kc) };
  const unsigned lV[2] = { (unsigned)(uintptr_t)(Vs[0] + vd * 32 + vc),
                           (unsigned)(uintptr_t)(Vs[1] + vd * 32 + vc) };
  const __bf16* Ksrc = Kp + ((size_t)bh * Sc + kr) * HDc + kc;   // + kb*HDc per tile
  const __bf16* Vsrc = Vt + ((size_t)bh * HDc + vd) * Sc + vc;   // + kb per tile
  __bf16* PsW = Ps + w * (16 * 32);

  // prologue: start DMA for tile 0
  async_ld16(lK[0], Ksrc);
  async_ld16(lV[0], Vsrc);

  for (int j = 0; j < ntiles; ++j) {
    const int kb  = j * 32;
    const int cur = j & 1;
    const bool more = (j + 1) < ntiles;
    if (more) {                                 // start DMA for tile j+1
      async_ld16(lK[cur ^ 1], Ksrc + (size_t)(kb + 32) * HDc);
      async_ld16(lV[cur ^ 1], Vsrc + (kb + 32));
    }
    if (more) wait_async2(); else wait_async0();   // tile j landed (in-order)
    __syncthreads();

    if (kb <= qw + 15) {                        // else fully masked for this wave
      const __bf16* Kc = Ks[cur];
      const __bf16* Vc = Vs[cur];

      // scores (16 q) x (32 keys), accumulated over d in two K=32 chunks
      v8f s0 = {}, s1 = {};
      #pragma unroll
      for (int f = 0; f < 2; ++f) {
        v16bf kf0 = ld_frag(Kc + f * 32, 64, 0);
        v16bf kf1 = ld_frag(Kc + f * 32, 64, 16);
        s0 = __builtin_amdgcn_wmma_f32_16x16x32_bf16(false, qf[f], false, kf0, (short)0, s0, false, false);
        s1 = __builtin_amdgcn_wmma_f32_16x16x32_bf16(false, qf[f], false, kf1, (short)0, s1, false, false);
      }

      const bool needmask = (kb + 31) > qw;
      float alpha[8];
      #pragma unroll
      for (int r = 0; r < 8; ++r) {
        float v0 = s0[r], v1 = s1[r];
        if (needmask) {
          const int qrow = qw + r + half * 8;
          const int kcol = kb + (lane & 15);
          if (kcol > qrow)      v0 = -1e30f;
          if (kcol + 16 > qrow) v1 = -1e30f;
        }
        const float mn = fmaxf(mrow[r], rowmax16(fmaxf(v0, v1)));
        alpha[r] = __expf(mrow[r] - mn);
        mrow[r] = mn;
        const float p0 = __expf(v0 - mn);
        const float p1 = __expf(v1 - mn);
        const int mr = r + half * 8;
        PsW[mr * 32 + (lane & 15)]      = f2bf(p0);
        PsW[mr * 32 + 16 + (lane & 15)] = f2bf(p1);
      }
      asm volatile("" ::: "memory");           // keep ds_load after ds_store (HW in-order)
      const v16bf pf = ld_frag(PsW, 32, 0);    // P as 16x32 A-fragment

      // row-sum(P) via WMMA with all-ones B: every column of D = rowsum.
      v8f rs = {};
      rs = __builtin_amdgcn_wmma_f32_16x16x32_bf16(false, pf, false, onesf, (short)0, rs, false, false);
      #pragma unroll
      for (int r = 0; r < 8; ++r) lrow[r] = lrow[r] * alpha[r] + rs[r];

      #pragma unroll
      for (int t = 0; t < 4; ++t) {
        const v16bf vf = ld_frag(Vc, 32, t * 16);  // V columns d = 16t..16t+15
        #pragma unroll
        for (int r = 0; r < 8; ++r) o[t][r] *= alpha[r];
        o[t] = __builtin_amdgcn_wmma_f32_16x16x32_bf16(false, pf, false, vf, (short)0, o[t], false, false);
      }
    }
    __syncthreads();                            // readers done before next DMA reuses buffer
  }

  #pragma unroll
  for (int r = 0; r < 8; ++r) lrow[r] = 1.0f / lrow[r];
  #pragma unroll
  for (int t = 0; t < 4; ++t)
    #pragma unroll
    for (int r = 0; r < 8; ++r) {
      const int srow = qw + r + half * 8;
      const int dcol = h * HDc + t * 16 + (lane & 15);
      Oout[(size_t)(b * Sc + srow) * Dc + dcol] = f2bf(o[t][r] * lrow[r]);
    }
}

// ---------------------------------------------------------------------------
extern "C" void kernel_launch(void* const* d_in, const int* in_sizes, int n_in,
                              void* d_out, int out_size, void* d_ws, size_t ws_size,
                              hipStream_t stream) {
  const float* q  = (const float*)d_in[0];
  const float* k  = (const float*)d_in[1];
  const float* v  = (const float*)d_in[2];
  const float* wq = (const float*)d_in[3];
  const float* wk = (const float*)d_in[4];
  const float* wv = (const float*)d_in[5];
  const float* wo = (const float*)d_in[6];

  const size_t per = (size_t)Bc * Hc * Sc * HDc;    // 8 Mi elements
  __bf16* Qp = (__bf16*)d_ws;                       // [B,H,S,HD], pre-scaled
  __bf16* Kp = Qp + per;                            // [B,H,S,HD]
  __bf16* Vt = Kp + per;                            // [B,H,HD,S]
  __bf16* Ao = Vt + per;                            // [B,S,D] attention output

  const dim3 gg(Dc / 128, (Bc * Sc) / 128);
  gemm_wmma<1, false><<<gg, 256, 0, stream>>>(q, wq, Qp, 0.125f);   // 1/sqrt(64)
  gemm_wmma<1, false><<<gg, 256, 0, stream>>>(k, wk, Kp, 1.0f);
  gemm_wmma<2, false><<<gg, 256, 0, stream>>>(v, wv, Vt, 1.0f);
  flash_attn<<<dim3(Sc / 128, Bc * Hc), 256, 0, stream>>>(Qp, Kp, Vt, Ao);
  gemm_wmma<0, true><<<gg, 256, 0, stream>>>(Ao, wo, d_out, 1.0f);
}
